// BCTransformer_50199577755952
// MI455X (gfx1250) — matrix operand order repaired and checked by reading, not compiled, not run on hardware
//
#include <hip/hip_runtime.h>
#include <hip/hip_bf16.h>
#include <math.h>

// ---------------------------------------------------------------------------
// CDNA5 (gfx1250) batched tiny-transformer, round 3.
// Each wave32 owns TWO 16-row tiles (M=32: 16 batch elems x 2 tokens) and runs
// all 4 encoder layers + head MLP with v_wmma_f32_16x16x32_f16. Both A-tiles
// share each B-fragment load (32 FLOP per L2 byte), and the A fragments are
// EXPLICITLY hoisted into registers (v16h a[NK]) before the n-tile loop so the
// inner loop is a pure B-stream + WMMA pipeline (round-2 codegen had regressed
// to re-gathering A from LDS per kt iteration).
// ---------------------------------------------------------------------------

typedef __attribute__((ext_vector_type(16))) _Float16 v16h;
typedef __attribute__((ext_vector_type(8)))  float    v8f;

#define DMODEL 128
#define FFDIM  256
#define NLAYER 4
#define HS 132            // h row stride (132 % 64 == 4 -> conflict-free column gathers)
#define SS 516            // scr row stride: cols 0..383 gemm out, 384..511 LN buffer, +4 pad
#define WAVES 2           // waves/block; LDS = 2 waves * 2 tiles * 41,472 B = 165,888 B
#define TFLOATS (16 * HS + 16 * SS)

__device__ __forceinline__ void lds_fence() {
  asm volatile("s_wait_dscnt 0" ::: "memory");
}

__device__ __forceinline__ float gelu_exact(float v) {
  return 0.5f * v * (1.0f + erff(v * 0.70710678118654752f));
}

// Dual-tile GEMM: Y{0,1}[16 x NOUT] = gelu?(X{0,1}[16 x K] @ Wfrag + bias).
// A fragments register-resident across the whole n-loop; one B-fragment load
// feeds two WMMAs (M-blocking for L2 bandwidth).
template <int NOUT, int K, int ACT, int XS, int YS>
__device__ __forceinline__ void gemm_pair(const float* __restrict__ X0,
                                          const float* __restrict__ X1,
                                          const _Float16* __restrict__ W,
                                          const float* __restrict__ bias,
                                          float* __restrict__ Y0,
                                          float* __restrict__ Y1, int lane) {
  constexpr int NK = K / 32;
  constexpr int NN = NOUT / 16;
  const int m     = lane & 15;
  const int abase = (lane & 16) ? 8 : 0;      // ISA 16-bit A 16x32 layout
  const int mb    = (lane & 16) ? 8 : 0;      // ISA 32-bit C/D 16x16 layout
  const float* xr0 = X0 + m * XS;
  const float* xr1 = X1 + m * XS;

  // ---- build all A fragments once, keep in registers --------------------
  v16h a0[NK], a1[NK];
#pragma unroll
  for (int kt = 0; kt < NK; ++kt) {
#pragma unroll
    for (int i = 0; i < 8; ++i) {
      int k = kt * 32 + abase + ((i < 4) ? (2 * i) : (16 + 2 * (i - 4)));
      a0[kt][2 * i]     = (_Float16)xr0[k];
      a0[kt][2 * i + 1] = (_Float16)xr0[k + 1];
      a1[kt][2 * i]     = (_Float16)xr1[k];
      a1[kt][2 * i + 1] = (_Float16)xr1[k + 1];
    }
  }

  // ---- n-tile loop: pure B stream + WMMA ---------------------------------
  for (int nt = 0; nt < NN; ++nt) {
    v8f c0 = {}, c1 = {};
#pragma unroll
    for (int kt = 0; kt < NK; ++kt) {
      v16h b = *(const v16h*)(W + ((size_t)(nt * NK + kt) * 32 + lane) * 16);
      c0 = __builtin_amdgcn_wmma_f32_16x16x32_f16(false, a0[kt], false, b,
                                                  (short)0, c0, false, false);
      c1 = __builtin_amdgcn_wmma_f32_16x16x32_f16(false, a1[kt], false, b,
                                                  (short)0, c1, false, false);
    }
    int n = nt * 16 + (lane & 15);
    float bv = bias[n];
#pragma unroll
    for (int r = 0; r < 8; ++r) {
      float v0 = c0[r] + bv;
      float v1 = c1[r] + bv;
      if (ACT) { v0 = gelu_exact(v0); v1 = gelu_exact(v1); }
      Y0[(mb + r) * YS + n] = v0;
      Y1[(mb + r) * YS + n] = v1;
    }
  }
}

// LayerNorm over 16 rows of 128 cols: 2 lanes per row, shfl_xor combine.
__device__ __forceinline__ void layernorm16(const float* __restrict__ X, int xs,
                                            const float* __restrict__ g,
                                            const float* __restrict__ b,
                                            float* __restrict__ Yd, int ys,
                                            int lane) {
  int row = lane >> 1, half = lane & 1;
  const float* xr = X + row * xs;
  float s = 0.f, s2 = 0.f;
  for (int d = half; d < DMODEL; d += 2) { float v = xr[d]; s += v; s2 += v * v; }
  s  += __shfl_xor(s, 1, 32);
  s2 += __shfl_xor(s2, 1, 32);
  float mean = s * (1.0f / DMODEL);
  float inv  = rsqrtf(s2 * (1.0f / DMODEL) - mean * mean + 1e-5f);
  float* yr = Yd + row * ys;
  for (int d = half; d < DMODEL; d += 2)
    yr[d] = (xr[d] - mean) * inv * g[d] + b[d];
}

// Repack fp32 weight w[Nout][K] (y = x @ w^T) into f16 WMMA B fragments.
__global__ void pack_wmma_b(const float* __restrict__ w, _Float16* __restrict__ dst,
                            int Nout, int K) {
  int t = blockIdx.x * blockDim.x + threadIdx.x;
  int numK = K >> 5, numN = Nout >> 4;
  int total = numN * numK * 32;
  if (t >= total) return;
  int lane = t & 31, frag = t >> 5;
  int kt = frag % numK, nt = frag / numK;
  _Float16* o = dst + (size_t)frag * 512 + (size_t)lane * 16;
  int n  = nt * 16 + (lane & 15);
  int kb = kt * 32 + ((lane & 16) ? 8 : 0);
#pragma unroll
  for (int i = 0; i < 16; ++i) o[i] = (_Float16)w[(size_t)n * K + kb + i + ((i >= 8) ? 8 : 0)];
}

struct P {
  const float *x, *w_in, *b_in, *g_in, *bt_in;
  const float *qkv_b, *out_b, *n1_g, *n1_b, *n2_g, *n2_b, *ff1_b, *ff2_b;
  const float *g_out, *bt_out, *h1_b, *h2_b, *h3_w, *h3_b;
  const _Float16 *qkvF, *outF, *ff1F, *ff2F, *h1F, *h2F;
  float* out;
  int nbatch;
};

__global__ __launch_bounds__(WAVES * 32) void encoder_kernel(P p) {
  __shared__ float smem[WAVES][2][TFLOATS];
  const int wave = threadIdx.x >> 5;
  const int lane = threadIdx.x & 31;
  float* h0   = smem[wave][0];
  float* scr0 = h0 + 16 * HS;
  float* h1   = smem[wave][1];
  float* scr1 = h1 + 16 * HS;
  float* hA[2]   = { h0, h1 };
  float* scrA[2] = { scr0, scr1 };

  const int pair  = blockIdx.x * WAVES + wave;   // covers 16 batch elems
  const int bbase = pair * 16;
  if (bbase >= p.nbatch) return;   // wave-uniform

  // ---- input projection + LN + positional encoding ----------------------
  for (int t = 0; t < 2; ++t) {
    float* h = hA[t];
    int tb = bbase + t * 8;
    for (int idx = lane; idx < 16 * DMODEL; idx += 32) {
      int r = idx >> 7, d = idx & 127;
      int b = tb + (r >> 1), s = r & 1;
      float x0 = p.x[b * 4 + s * 2 + 0];
      float x1 = p.x[b * 4 + s * 2 + 1];
      h[r * HS + d] = (x0 * p.w_in[d * 2 + 0] + x1 * p.w_in[d * 2 + 1] + p.b_in[d])
                      * 11.313708498984761f;   // sqrt(128)
    }
  }
  lds_fence();
  for (int t = 0; t < 2; ++t)
    layernorm16(hA[t], HS, p.g_in, p.bt_in, scrA[t] + 384, SS, lane);
  lds_fence();
  for (int t = 0; t < 2; ++t) {
    float* h = hA[t]; float* scr = scrA[t];
    for (int idx = lane; idx < 16 * DMODEL; idx += 32) {
      int r = idx >> 7, d = idx & 127;
      int s = r & 1;
      float div = expf((float)(d & ~1) * (-0.0719557841560639f)); // ln(1e4)/128
      float ang = (float)s * div;
      float pe  = (d & 1) ? cosf(ang) : sinf(ang);
      h[r * HS + d] = scr[r * SS + 384 + d] + pe;
    }
  }
  lds_fence();

  // ---- 4 pre-norm encoder layers ----------------------------------------
  for (int l = 0; l < NLAYER; ++l) {
    const _Float16* qkvW = p.qkvF + (size_t)l * 96 * 512;
    const _Float16* outW = p.outF + (size_t)l * 32 * 512;
    const _Float16* ff1W = p.ff1F + (size_t)l * 64 * 512;
    const _Float16* ff2W = p.ff2F + (size_t)l * 64 * 512;

    for (int t = 0; t < 2; ++t)
      layernorm16(hA[t], HS, p.n1_g + l * DMODEL, p.n1_b + l * DMODEL,
                  scrA[t] + 384, SS, lane);
    lds_fence();
    // qkv: scr[.,0:384] = y @ qkv_w^T + b  (24 n-tiles x 4 k-tiles, B shared)
    gemm_pair<384, 128, 0, SS, SS>(scr0 + 384, scr1 + 384, qkvW,
                                   p.qkv_b + l * 384, scr0, scr1, lane);
    lds_fence();

    // attention: S=2, 4 heads of 32; lane = (batch-in-tile i)*4 + head.
    for (int t = 0; t < 2; ++t) {
      float* scr = scrA[t];
      int i = lane >> 2, hh = lane & 3;
      const float* q0 = scr + (2 * i) * SS + hh * 32;
      const float* q1 = scr + (2 * i + 1) * SS + hh * 32;
      const float* k0 = q0 + 128; const float* k1 = q1 + 128;
      const float* v0 = q0 + 256; const float* v1 = q1 + 256;
      float a00 = 0, a01 = 0, a10 = 0, a11 = 0;
#pragma unroll 4
      for (int d = 0; d < 32; ++d) {
        float qa = q0[d], qb = q1[d], ka = k0[d], kb = k1[d];
        a00 += qa * ka; a01 += qa * kb; a10 += qb * ka; a11 += qb * kb;
      }
      const float sc = 0.17677669529663687f;  // 1/sqrt(32)
      a00 *= sc; a01 *= sc; a10 *= sc; a11 *= sc;
      float m0 = fmaxf(a00, a01), m1 = fmaxf(a10, a11);
      float e00 = expf(a00 - m0), e01 = expf(a01 - m0);
      float e10 = expf(a10 - m1), e11 = expf(a11 - m1);
      float r0 = 1.0f / (e00 + e01), r1 = 1.0f / (e10 + e11);
      float w00 = e00 * r0, w01 = e01 * r0, w10 = e10 * r1, w11 = e11 * r1;
      float* o0 = scr + (2 * i) * SS + hh * 32;      // overwrite q slice
      float* o1 = scr + (2 * i + 1) * SS + hh * 32;
#pragma unroll 4
      for (int d = 0; d < 32; ++d) {
        float va = v0[d], vb = v1[d];
        o0[d] = w00 * va + w01 * vb;
        o1[d] = w10 * va + w11 * vb;
      }
    }
    lds_fence();
    // out proj: scr[.,128:256] = o @ out_w^T + b ; residual into h
    gemm_pair<128, 128, 0, SS, SS>(scr0, scr1, outW, p.out_b + l * DMODEL,
                                   scr0 + 128, scr1 + 128, lane);
    lds_fence();
    for (int t = 0; t < 2; ++t) {
      float* h = hA[t]; float* scr = scrA[t];
      for (int idx = lane; idx < 16 * DMODEL; idx += 32) {
        int r = idx >> 7, d = idx & 127;
        h[r * HS + d] += scr[r * SS + 128 + d];
      }
    }
    lds_fence();

    for (int t = 0; t < 2; ++t)
      layernorm16(hA[t], HS, p.n2_g + l * DMODEL, p.n2_b + l * DMODEL,
                  scrA[t] + 384, SS, lane);
    lds_fence();
    // ff1 + exact gelu: scr[.,0:256]
    gemm_pair<256, 128, 1, SS, SS>(scr0 + 384, scr1 + 384, ff1W,
                                   p.ff1_b + l * FFDIM, scr0, scr1, lane);
    lds_fence();
    // ff2: scr[.,256:384]
    gemm_pair<128, 256, 0, SS, SS>(scr0, scr1, ff2W, p.ff2_b + l * DMODEL,
                                   scr0 + 256, scr1 + 256, lane);
    lds_fence();
    for (int t = 0; t < 2; ++t) {
      float* h = hA[t]; float* scr = scrA[t];
      for (int idx = lane; idx < 16 * DMODEL; idx += 32) {
        int r = idx >> 7, d = idx & 127;
        h[r * HS + d] += scr[r * SS + 256 + d];
      }
    }
    lds_fence();
  }

  // ---- final LN + token-mean pool ---------------------------------------
  for (int t = 0; t < 2; ++t)
    layernorm16(hA[t], HS, p.g_out, p.bt_out, scrA[t] + 384, SS, lane);
  lds_fence();
  for (int t = 0; t < 2; ++t) {
    float* h = hA[t]; float* scr = scrA[t];
    for (int idx = lane; idx < 16 * DMODEL; idx += 32) {
      int r = idx >> 7, d = idx & 127;
      float v = 0.f;
      if (r < 8)
        v = 0.5f * (scr[(2 * r) * SS + 384 + d] + scr[(2 * r + 1) * SS + 384 + d]);
      h[r * HS + d] = v;   // rows 8..15 zeroed -> safe WMMA padding
    }
  }
  lds_fence();
  // head MLP on padded 16-row tiles (B fragments shared across the pair)
  gemm_pair<128, 128, 1, HS, SS>(h0, h1, p.h1F, p.h1_b, scr0, scr1, lane);
  lds_fence();
  gemm_pair<64, 128, 1, SS, SS>(scr0, scr1, p.h2F, p.h2_b,
                                scr0 + 128, scr1 + 128, lane);
  lds_fence();
  for (int t = 0; t < 2; ++t) {
    if (lane < 8) {
      float acc = p.h3_b[0];
      const float* pr = scrA[t] + lane * SS + 128;
#pragma unroll 8
      for (int j = 0; j < 64; ++j) acc += pr[j] * p.h3_w[j];
      acc = 3.0f * tanhf(acc);
      acc = fminf(3.0f, fmaxf(-3.0f, acc));
      p.out[bbase + t * 8 + lane] = acc;
    }
  }
}

extern "C" void kernel_launch(void* const* d_in, const int* in_sizes, int n_in,
                              void* d_out, int out_size, void* d_ws, size_t ws_size,
                              hipStream_t stream) {
  const float* x     = (const float*)d_in[0];
  const float* w_in  = (const float*)d_in[1];
  const float* b_in  = (const float*)d_in[2];
  const float* g_in  = (const float*)d_in[3];
  const float* bt_in = (const float*)d_in[4];
  const float* qkv_w = (const float*)d_in[5];
  const float* qkv_b = (const float*)d_in[6];
  const float* out_w = (const float*)d_in[7];
  const float* out_b = (const float*)d_in[8];
  const float* n1_g  = (const float*)d_in[9];
  const float* n1_b  = (const float*)d_in[10];
  const float* n2_g  = (const float*)d_in[11];
  const float* n2_b  = (const float*)d_in[12];
  const float* ff1_w = (const float*)d_in[13];
  const float* ff1_b = (const float*)d_in[14];
  const float* ff2_w = (const float*)d_in[15];
  const float* ff2_b = (const float*)d_in[16];
  const float* g_out  = (const float*)d_in[17];
  const float* bt_out = (const float*)d_in[18];
  const float* h1_w = (const float*)d_in[19];
  const float* h1_b = (const float*)d_in[20];
  const float* h2_w = (const float*)d_in[21];
  const float* h2_b = (const float*)d_in[22];
  const float* h3_w = (const float*)d_in[23];
  const float* h3_b = (const float*)d_in[24];

  // Workspace: f16 B-fragment-packed weights (~1.05 MB, L2-resident).
  _Float16* qkvF = (_Float16*)d_ws;                 // 4 * 96 * 512
  _Float16* outF = qkvF + (size_t)4 * 96 * 512;     // 4 * 32 * 512
  _Float16* ff1F = outF + (size_t)4 * 32 * 512;     // 4 * 64 * 512
  _Float16* ff2F = ff1F + (size_t)4 * 64 * 512;     // 4 * 64 * 512
  _Float16* h1F  = ff2F + (size_t)4 * 64 * 512;     // 32 * 512
  _Float16* h2F  = h1F  + (size_t)32 * 512;         // 16 * 512

  for (int l = 0; l < 4; ++l) {
    pack_wmma_b<<<(24 * 4 * 32 + 255) / 256, 256, 0, stream>>>(
        qkv_w + (size_t)l * 384 * 128, qkvF + (size_t)l * 96 * 512, 384, 128);
    pack_wmma_b<<<(8 * 4 * 32 + 255) / 256, 256, 0, stream>>>(
        out_w + (size_t)l * 128 * 128, outF + (size_t)l * 32 * 512, 128, 128);
    pack_wmma_b<<<(16 * 4 * 32 + 255) / 256, 256, 0, stream>>>(
        ff1_w + (size_t)l * 256 * 128, ff1F + (size_t)l * 64 * 512, 256, 128);
    pack_wmma_b<<<(8 * 8 * 32 + 255) / 256, 256, 0, stream>>>(
        ff2_w + (size_t)l * 128 * 256, ff2F + (size_t)l * 64 * 512, 128, 256);
  }
  pack_wmma_b<<<(32 * 32 + 255) / 256, 256, 0, stream>>>(h1_w, h1F, 128, 128);
  pack_wmma_b<<<(16 * 32 + 255) / 256, 256, 0, stream>>>(h2_w, h2F, 64, 128);

  int B = in_sizes[0] / 4;
  P p = { x, w_in, b_in, g_in, bt_in,
          qkv_b, out_b, n1_g, n1_b, n2_g, n2_b, ff1_b, ff2_b,
          g_out, bt_out, h1_b, h2_b, h3_w, h3_b,
          qkvF, outF, ff1F, ff2F, h1F, h2F,
          (float*)d_out, B };

  int pairs  = (B + 15) / 16;                  // 32 rows (2 tiles) per wave
  int blocks = (pairs + WAVES - 1) / WAVES;    // B=131072 -> 4096 blocks
  encoder_kernel<<<blocks, WAVES * 32, 0, stream>>>(p);
}
